// MultiHeadedLatentAttention_44427141710495
// MI455X (gfx1250) — matrix operand order repaired and checked by reading, not compiled, not run on hardware
//
#include <hip/hip_runtime.h>

#define B_   2
#define S_   2048
#define DIM_ 2048
#define H_   16
#define DH_  128
#define LKV_ 512
#define LQ_  1024
#define M_   (B_*S_)

typedef _Float16 v16h __attribute__((ext_vector_type(16)));
typedef _Float16 v8h  __attribute__((ext_vector_type(8)));
typedef _Float16 v4h  __attribute__((ext_vector_type(4)));
typedef float    v8f  __attribute__((ext_vector_type(8)));
typedef unsigned int v4u __attribute__((ext_vector_type(4)));
typedef int      v4i  __attribute__((ext_vector_type(4)));
typedef int      v8i  __attribute__((ext_vector_type(8)));

union Frag16 { v16h v; v8h h[2]; };

__device__ __forceinline__ v8f wmma16(v16h a, v16h b, v8f c) {
  // D(16x16 f32) = A(16x32 f16) * B(32x16 f16) + C
  return __builtin_amdgcn_wmma_f32_16x16x32_f16(false, a, false, b, (short)0, c, false, false);
}

// ---------------------------------------------------------------------------
// TDM 2-D tile load (f16 elements) into LDS. Descriptor per CDNA5 ISA:
//  group0: [1:0]count=1 | [63:32]lds_addr | [120:64]global_addr | [127:126]type=2
//  group1: d0 = data_size(2B) + pad config; tensor dims == tile dims (no OOB);
//          d5 = tensor_dim0_stride (elements). groups 2/3 zero (2-D tensor).
// d0_flags: K/Q rows 512B: pad_en|interval=6(128 dw)|amount=3(4 dw) = 0x07910000
//           V   rows  64B: pad_en|interval=3(16 dw)|amount=3(4 dw)  = 0x06D10000
// This toolchain declares the 6-arg builtin: (v4u, v8i, v4i, v4i, v8i, i32 cpol).
// ---------------------------------------------------------------------------
__device__ __forceinline__ void tdm_load_2d_f16(unsigned lds_byte_addr, const _Float16* gptr,
                                                unsigned tile_d0, unsigned tile_d1,
                                                unsigned stride_elems, unsigned d0_flags)
{
  unsigned long long ga = (unsigned long long)(uintptr_t)gptr;
  v4u g0 = { 1u, lds_byte_addr, (unsigned)ga,
             (unsigned)((ga >> 32) & 0x1ffffffu) | 0x80000000u };
  v8i g1 = { (int)d0_flags,
             (int)(tile_d0 << 16),        // tensor_dim0[15:0]
             (int)(tile_d1 << 16),        // tensor_dim0[31:16]=0 | tensor_dim1[15:0]
             (int)(tile_d0 << 16),        // tensor_dim1[31:16]=0 | tile_dim0
             (int)(tile_d1 & 0xffffu),    // tile_dim1 | tile_dim2=0
             (int)stride_elems,           // tensor_dim0_stride[31:0]
             0, 0 };
  v4i gz4 = { 0, 0, 0, 0 };
  v8i gz8 = { 0, 0, 0, 0, 0, 0, 0, 0 };
  __builtin_amdgcn_tensor_load_to_lds(g0, g1, gz4, gz4, gz8, 0);
}

// ---------------------------------------------------------------------------
// fp32 -> fp16 conversion, 4 elements / thread
// ---------------------------------------------------------------------------
__global__ __launch_bounds__(256)
void cvt_f32_f16_kernel(const float* __restrict__ in, _Float16* __restrict__ out, int n4)
{
  int i = blockIdx.x * 256 + threadIdx.x;
  if (i >= n4) return;
  float4 v = ((const float4*)in)[i];
  v4h o = { (_Float16)v.x, (_Float16)v.y, (_Float16)v.z, (_Float16)v.w };
  ((v4h*)out)[i] = o;
}

// ---------------------------------------------------------------------------
// f16 GEMM: acc_f32 = X(f16)[m,k] * W(f16)[n,k]^T  (+ bias_f32[n])
// OMODE 0: f16 out, scatter out[m*row_stride + (n>>7)*head_stride + (n&127)]
// OMODE 1: f16 out, V-transposed: out[((b*H + h)*128 + d)*S + s]
// OMODE 2: f32 out, plain: out[m*row_stride + n]
// Block 256 thr (8 waves), tile 64(M) x 128(N), wave 32x32, K slab 32.
// ---------------------------------------------------------------------------
template<int OMODE>
__global__ __launch_bounds__(256)
void gemm16_kernel(const _Float16* __restrict__ X, const _Float16* __restrict__ W,
                   const float* __restrict__ bias, void* __restrict__ outv,
                   int K, int row_stride, int head_stride)
{
  __shared__ __align__(16) _Float16 As[64][40];
  __shared__ __align__(16) _Float16 Bs[128][40];

  const int t     = threadIdx.x;
  const int wave  = t >> 5, lane = t & 31;
  const int wm    = wave >> 2, wn = wave & 3;      // 2 x 4 wave grid
  const int mrow  = lane & 15;
  const int abase = (lane >> 4) << 3;              // A frag K chunk base (0 / 8)
  const int ko    = (lane >> 4) << 3;              // B frag K base (0 / 8)
  const int half8 = (lane >> 4) << 3;              // C layout row offset

  v8f acc00 = {}, acc01 = {}, acc10 = {}, acc11 = {};

  const _Float16* Xp = X + (size_t)blockIdx.x * 64  * K;
  const _Float16* Wp = W + (size_t)blockIdx.y * 128 * K;

  const int arow = t >> 2, ac8 = (t & 3) << 3;     // A: 64 rows x 4 v8h

  for (int kt = 0; kt < K; kt += 32) {
    __syncthreads();
    *(v8h*)&As[arow][ac8] = *(const v8h*)&Xp[(size_t)arow * K + kt + ac8];
    {
      int row = t >> 2, c8 = (t & 3) << 3;         // B: 128 rows x 4 v8h (2/thr)
      *(v8h*)&Bs[row][c8] = *(const v8h*)&Wp[(size_t)row * K + kt + c8];
      int idx = t + 256; row = idx >> 2; c8 = (idx & 3) << 3;
      *(v8h*)&Bs[row][c8] = *(const v8h*)&Wp[(size_t)row * K + kt + c8];
    }
    __syncthreads();

    Frag16 a0, a1, b0, b1;
    a0.h[0] = *(const v8h*)&As[wm*32 +      mrow][abase];
    a0.h[1] = *(const v8h*)&As[wm*32 +      mrow][abase + 16];
    a1.h[0] = *(const v8h*)&As[wm*32 + 16 + mrow][abase];
    a1.h[1] = *(const v8h*)&As[wm*32 + 16 + mrow][abase + 16];
    b0.h[0] = *(const v8h*)&Bs[wn*32 +      mrow][ko];
    b0.h[1] = *(const v8h*)&Bs[wn*32 +      mrow][ko + 16];
    b1.h[0] = *(const v8h*)&Bs[wn*32 + 16 + mrow][ko];
    b1.h[1] = *(const v8h*)&Bs[wn*32 + 16 + mrow][ko + 16];

    acc00 = wmma16(a0.v, b0.v, acc00);
    acc01 = wmma16(a0.v, b1.v, acc01);
    acc10 = wmma16(a1.v, b0.v, acc10);
    acc11 = wmma16(a1.v, b1.v, acc11);
  }

  const int colh = lane & 15;
  #pragma unroll
  for (int i = 0; i < 2; ++i) {
    #pragma unroll
    for (int j = 0; j < 2; ++j) {
      v8f a = (i == 0) ? (j == 0 ? acc00 : acc01) : (j == 0 ? acc10 : acc11);
      int col = blockIdx.y * 128 + wn * 32 + j * 16 + colh;
      float bv = bias ? bias[col] : 0.0f;
      #pragma unroll
      for (int r = 0; r < 8; ++r) {
        int rowg = blockIdx.x * 64 + wm * 32 + i * 16 + r + half8;
        float val = a[r] + bv;
        if (OMODE == 0) {
          int hcol = (col >> 7) * head_stride + (col & 127);
          ((_Float16*)outv)[(size_t)rowg * row_stride + hcol] = (_Float16)val;
        } else if (OMODE == 1) {
          // rowg = b*S + s ; col = h*128 + d ; out[((b*H+h)*128+d)*S + s]
          size_t idx = ((((size_t)(rowg >> 11) * H_) + (col >> 7)) * 128 + (col & 127)) * S_
                       + (rowg & (S_ - 1));
          ((_Float16*)outv)[idx] = (_Float16)val;
        } else {
          ((float*)outv)[(size_t)rowg * row_stride + col] = val;
        }
      }
    }
  }
}

// ---------------------------------------------------------------------------
// RoPE on the rotary half (offset DH_ of each 256-wide head), f16 in-place.
// ---------------------------------------------------------------------------
__global__ __launch_bounds__(256)
void rope16_kernel(_Float16* __restrict__ qf, _Float16* __restrict__ kf,
                   const float* __restrict__ fcos, const float* __restrict__ fsin)
{
  int idx = blockIdx.x * 256 + threadIdx.x;        // B*S*H*64 exact
  int i = idx & 63;
  int h = (idx >> 6) & (H_ - 1);
  int s = (idx >> 10) & (S_ - 1);
  int b = idx >> 21;
  float c  = fcos[s * 64 + i];
  float sn = fsin[s * 64 + i];
  size_t base = ((((size_t)b * S_ + s) * H_ + h) << 8) + DH_ + 2 * i;
  float ar = (float)qf[base], ai = (float)qf[base + 1];
  qf[base]     = (_Float16)(ar * c - ai * sn);
  qf[base + 1] = (_Float16)(ar * sn + ai * c);
  ar = (float)kf[base]; ai = (float)kf[base + 1];
  kf[base]     = (_Float16)(ar * c - ai * sn);
  kf[base + 1] = (_Float16)(ar * sn + ai * c);
}

// ---------------------------------------------------------------------------
// Flash attention (f16 WMMA, f32 accum): one (b, h, 64-query block) per WG.
// 4 waves x 16 queries; key tile 32; d_qk=256, d_v=128. Q/K/V tiles staged by
// the Tensor Data Mover (wave 0 issues, TENSORcnt wait, then WG barrier).
// V is pre-transposed [b][h][d][s] so P@V B-fragments are contiguous.
// ---------------------------------------------------------------------------
__global__ __launch_bounds__(128)
void mla_flash16_kernel(const _Float16* __restrict__ Qf, const _Float16* __restrict__ Kf,
                        const _Float16* __restrict__ Vt, _Float16* __restrict__ Of)
{
  __shared__ __align__(16) _Float16 Qs[64][264];
  __shared__ __align__(16) _Float16 Ks[32][264];
  __shared__ __align__(16) _Float16 Vs[128][40];   // [d][key]
  __shared__ __align__(16) _Float16 Ps[4][16][40];

  const int t     = threadIdx.x;
  const int wave  = t >> 5, lane = t & 31;
  const int b     = blockIdx.z, h = blockIdx.y;
  const int qb    = blockIdx.x * 64;
  const int mrow  = lane & 15;
  const int abase = (lane >> 4) << 3;
  const int ko    = (lane >> 4) << 3;
  const int half8 = (lane >> 4) << 3;
  const float scale = 0.08838834764831845f;        // 1/sqrt(128)

  const unsigned ldsQ = (unsigned)(uintptr_t)&Qs[0][0];
  const unsigned ldsK = (unsigned)(uintptr_t)&Ks[0][0];
  const unsigned ldsV = (unsigned)(uintptr_t)&Vs[0][0];

  // Stage Q tile (64 x 256, row stride H*256) via TDM; padded to [64][264].
  if (wave == 0) {
    tdm_load_2d_f16(ldsQ, Qf + ((((size_t)b * S_ + qb) * H_ + h) << 8),
                    256, 64, H_ * 256, 0x07910000u);
  }

  v8f o[8] = {};
  float m_run[8], l_run[8];
  #pragma unroll
  for (int r = 0; r < 8; ++r) { m_run[r] = -1e30f; l_run[r] = 0.0f; }

  for (int kt = 0; kt < S_; kt += 32) {
    __syncthreads();                               // tiles free to overwrite
    if (wave == 0) {
      tdm_load_2d_f16(ldsK, Kf + ((((size_t)b * S_ + kt) * H_ + h) << 8),
                      256, 32, H_ * 256, 0x07910000u);
      tdm_load_2d_f16(ldsV, Vt + ((size_t)b * H_ + h) * 128 * S_ + kt,
                      32, 128, S_, 0x06D10000u);
      __builtin_amdgcn_s_wait_tensorcnt(0);        // also covers Q on iter 0
    }
    __syncthreads();                               // publish tiles to all waves

    // scores: 16 queries x 32 keys, contraction 256 -> 8 WMMA K-steps
    v8f s0 = {}, s1 = {};
    const int q0 = wave * 16 + mrow;
    #pragma unroll
    for (int ks = 0; ks < 256; ks += 32) {
      Frag16 a, bb0, bb1;
      a.h[0]   = *(const v8h*)&Qs[q0][ks + abase];
      a.h[1]   = *(const v8h*)&Qs[q0][ks + abase + 16];
      bb0.h[0] = *(const v8h*)&Ks[mrow][ks + ko];
      bb0.h[1] = *(const v8h*)&Ks[mrow][ks + ko + 16];
      bb1.h[0] = *(const v8h*)&Ks[16 + mrow][ks + ko];
      bb1.h[1] = *(const v8h*)&Ks[16 + mrow][ks + ko + 16];
      s0 = wmma16(a.v, bb0.v, s0);
      s1 = wmma16(a.v, bb1.v, s1);
    }

    // online softmax; row r lives in this lane half's 16 lanes (C layout)
    #pragma unroll
    for (int r = 0; r < 8; ++r) {
      float e0 = s0[r] * scale;
      float e1 = s1[r] * scale;
      float mx = fmaxf(e0, e1);
      mx = fmaxf(mx, __shfl_xor(mx, 1, 32));
      mx = fmaxf(mx, __shfl_xor(mx, 2, 32));
      mx = fmaxf(mx, __shfl_xor(mx, 4, 32));
      mx = fmaxf(mx, __shfl_xor(mx, 8, 32));
      float mnew = fmaxf(m_run[r], mx);
      float corr = __expf(m_run[r] - mnew);
      m_run[r] = mnew;
      float p0 = __expf(e0 - mnew);
      float p1 = __expf(e1 - mnew);
      float rs = p0 + p1;
      rs += __shfl_xor(rs, 1, 32);
      rs += __shfl_xor(rs, 2, 32);
      rs += __shfl_xor(rs, 4, 32);
      rs += __shfl_xor(rs, 8, 32);
      l_run[r] = l_run[r] * corr + rs;
      #pragma unroll
      for (int j = 0; j < 8; ++j) o[j][r] *= corr;
      Ps[wave][r + half8][mrow]      = (_Float16)p0;   // keys 0..15
      Ps[wave][r + half8][16 + mrow] = (_Float16)p1;   // keys 16..31
    }

    // O += P(16x32) @ V(32x128): single K=32 WMMA step, 8 N-tiles.
    // Per-wave LDS RAW is in-order -> no barrier needed for Ps.
    Frag16 ap;
    ap.h[0] = *(const v8h*)&Ps[wave][mrow][abase];
    ap.h[1] = *(const v8h*)&Ps[wave][mrow][abase + 16];
    #pragma unroll
    for (int j = 0; j < 8; ++j) {
      Frag16 bv;
      bv.h[0] = *(const v8h*)&Vs[j * 16 + mrow][ko];
      bv.h[1] = *(const v8h*)&Vs[j * 16 + mrow][ko + 16];
      o[j] = wmma16(ap.v, bv.v, o[j]);
    }
  }

  #pragma unroll
  for (int r = 0; r < 8; ++r) {
    int row = wave * 16 + r + half8;
    float inv = 1.0f / l_run[r];
    size_t base = (((size_t)b * S_ + qb + row) * H_ + h) << 7;
    #pragma unroll
    for (int j = 0; j < 8; ++j)
      Of[base + j * 16 + mrow] = (_Float16)(o[j][r] * inv);
  }
}

// ---------------------------------------------------------------------------
extern "C" void kernel_launch(void* const* d_in, const int* in_sizes, int n_in,
                              void* d_out, int out_size, void* d_ws, size_t ws_size,
                              hipStream_t stream)
{
  (void)in_sizes; (void)n_in; (void)out_size; (void)ws_size;
  const float* x     = (const float*)d_in[0];
  const float* fcos  = (const float*)d_in[1];
  const float* fsin  = (const float*)d_in[2];
  const float* w_lq  = (const float*)d_in[3];
  const float* w_lkv = (const float*)d_in[4];
  const float* w_q   = (const float*)d_in[5];
  const float* w_k   = (const float*)d_in[6];
  const float* w_v   = (const float*)d_in[7];
  const float* w_qr  = (const float*)d_in[8];
  const float* b_qr  = (const float*)d_in[9];
  const float* w_kr  = (const float*)d_in[10];
  const float* b_kr  = (const float*)d_in[11];
  const float* w_o   = (const float*)d_in[12];
  const float* b_o   = (const float*)d_in[13];

  _Float16* p = (_Float16*)d_ws;
  _Float16* xh    = p; p += (size_t)M_ * DIM_;
  _Float16* wlqh  = p; p += (size_t)LQ_ * DIM_;
  _Float16* wlkvh = p; p += (size_t)LKV_ * DIM_;
  _Float16* wqh   = p; p += (size_t)(H_*DH_) * LQ_;
  _Float16* wqrh  = p; p += (size_t)(H_*DH_) * LQ_;
  _Float16* wkh   = p; p += (size_t)(H_*DH_) * LKV_;
  _Float16* wvh   = p; p += (size_t)(H_*DH_) * LKV_;
  _Float16* wkrh  = p; p += (size_t)(H_*DH_) * DIM_;
  _Float16* woh   = p; p += (size_t)DIM_ * (H_*DH_);
  _Float16* cqh   = p; p += (size_t)M_ * LQ_;
  _Float16* ckvh  = p; p += (size_t)M_ * LKV_;
  _Float16* qfh   = p; p += (size_t)M_ * H_ * 2 * DH_;   // [B,S,H,256]
  _Float16* kfh   = p; p += (size_t)M_ * H_ * 2 * DH_;   // [B,S,H,256]
  _Float16* vvT   = p; p += (size_t)M_ * H_ * DH_;       // [B,H,128,S]
  _Float16* aoh   = p; p += (size_t)M_ * H_ * DH_;       // [M, H*DH]

  dim3 blk(256);
  auto cvt = [&](const float* src, _Float16* dst, size_t n) {
    int n4 = (int)(n / 4);
    cvt_f32_f16_kernel<<<(n4 + 255) / 256, blk, 0, stream>>>(src, dst, n4);
  };
  cvt(x,     xh,    (size_t)M_ * DIM_);
  cvt(w_lq,  wlqh,  (size_t)LQ_ * DIM_);
  cvt(w_lkv, wlkvh, (size_t)LKV_ * DIM_);
  cvt(w_q,   wqh,   (size_t)(H_*DH_) * LQ_);
  cvt(w_qr,  wqrh,  (size_t)(H_*DH_) * LQ_);
  cvt(w_k,   wkh,   (size_t)(H_*DH_) * LKV_);
  cvt(w_v,   wvh,   (size_t)(H_*DH_) * LKV_);
  cvt(w_kr,  wkrh,  (size_t)(H_*DH_) * DIM_);
  cvt(w_o,   woh,   (size_t)DIM_ * (H_*DH_));

  // Projections (f16 in, f32 accum, f16 out)
  gemm16_kernel<0><<<dim3(M_/64, LQ_/128),  blk, 0, stream>>>(xh,   wlqh,  nullptr, cqh,        DIM_,    LQ_,  128);
  gemm16_kernel<0><<<dim3(M_/64, LKV_/128), blk, 0, stream>>>(xh,   wlkvh, nullptr, ckvh,       DIM_,    LKV_, 128);
  gemm16_kernel<0><<<dim3(M_/64, (H_*DH_)/128), blk, 0, stream>>>(cqh,  wqh,  nullptr, qfh,       LQ_,  H_*2*DH_, 256);
  gemm16_kernel<0><<<dim3(M_/64, (H_*DH_)/128), blk, 0, stream>>>(cqh,  wqrh, b_qr,    qfh + DH_, LQ_,  H_*2*DH_, 256);
  gemm16_kernel<0><<<dim3(M_/64, (H_*DH_)/128), blk, 0, stream>>>(ckvh, wkh,  nullptr, kfh,       LKV_, H_*2*DH_, 256);
  gemm16_kernel<0><<<dim3(M_/64, (H_*DH_)/128), blk, 0, stream>>>(xh,   wkrh, b_kr,    kfh + DH_, DIM_, H_*2*DH_, 256);
  gemm16_kernel<1><<<dim3(M_/64, (H_*DH_)/128), blk, 0, stream>>>(ckvh, wvh,  nullptr, vvT,       LKV_, 0, 0);

  rope16_kernel<<<(B_*S_*H_*(DH_/2))/256, blk, 0, stream>>>(qfh, kfh, fcos, fsin);

  mla_flash16_kernel<<<dim3(S_/64, H_, B_), dim3(128), 0, stream>>>(qfh, kfh, vvT, aoh);

  gemm16_kernel<2><<<dim3(M_/64, DIM_/128), blk, 0, stream>>>(aoh, woh, b_o, (float*)d_out, H_*DH_, DIM_, 128);
}
